// GroupQueryAttention_43645457662368
// MI455X (gfx1250) — compile-verified, matrix operand order checked by reading, not run on hardware
//
#include <hip/hip_runtime.h>

// ---------------------------------------------------------------------------
// GQA forward for MI455X (gfx1250, wave32, WMMA, async-to-LDS).
//   N_EMBD=2048, N_HEAD=16, N_GROUP=4, D_HEAD=128, HG=4, B=4, S=2048
// Pipeline (bf16 intermediates, f32 accumulation everywhere):
//   1. gemm_wmma<1,float>: Qp = (q@Wq+bq)*softmax_scale -> bf16 [B,G,HG,S,D]
//   2. gemm_wmma<2,float>: Kp = k@Wk+bk                 -> bf16 [B,G,S,D]
//   3. gemm_wmma<2,float>: Vp = v@Wv+bv                 -> bf16 [B,G,S,D]
//   4. gqa_attn          : flash-style attention        -> ctx bf16 [B,S,E]
//   5. gemm_wmma<0,__bf16>: out = ctx@Wo+bo             -> f32  [B,S,E]
// GEMM: 128x256 block, 8 waves @ 64x64 (16 WMMA / 8 fragments per K-chunk),
// double-buffered LDS, async b128 staging for bf16 A, conflict-aware W
// transpose (k index lane-fast).
// ---------------------------------------------------------------------------

typedef __attribute__((ext_vector_type(16))) __bf16 v16bf;
typedef __attribute__((ext_vector_type(8)))  __bf16 v8bf;
typedef __attribute__((ext_vector_type(8)))  float  v8f;

#define DEV __device__ __forceinline__

constexpr int kB  = 4;
constexpr int kS  = 2048;
constexpr int kE  = 2048;
constexpr int kG  = 4;
constexpr int kHG = 4;
constexpr int kD  = 128;

DEV v8f wmma_bf16(v16bf a, v16bf b, v8f c) {
  // D = A(16x32 bf16) x B(32x16 bf16) + C(16x16 f32)
  return __builtin_amdgcn_wmma_f32_16x16x32_bf16(
      /*neg_a=*/false, a, /*neg_b=*/false, b,
      /*c_mod=*/(short)0, c, /*reuse_a=*/false, /*reuse_b=*/false);
}

DEV v8f vzero8() {
  v8f z = {0.f, 0.f, 0.f, 0.f, 0.f, 0.f, 0.f, 0.f};
  return z;
}

// Build a 16-bit A/B fragment from a row of 32 contiguous K values (any AS).
// ISA layout (16-bit A 16x32): lane half (l>>4) selects K groups:
//   elements 0..7  <-> K = khalf*8 + 0..7 ; 8..15 <-> K = 16 + khalf*8 + 0..7
DEV v16bf frag_ld(const __bf16* rowptr, int khalf) {
  v8bf lo = *(const v8bf*)(rowptr + khalf * 8);
  v8bf hi = *(const v8bf*)(rowptr + 16 + khalf * 8);
  return __builtin_shufflevector(lo, hi, 0, 1, 2, 3, 4, 5, 6, 7,
                                 8, 9, 10, 11, 12, 13, 14, 15);
}

// Async global->LDS 16-byte copy (per enabled lane), tracked by ASYNCcnt.
DEV void async_copy_b128(unsigned lds_off, const void* gaddr) {
  asm volatile("global_load_async_to_lds_b128 %0, %1, off"
               :: "v"(lds_off), "v"(gaddr) : "memory");
}
DEV void wait_async0() {
  asm volatile("s_wait_asynccnt 0x0" ::: "memory");
}

// ---------------------------------------------------------------------------
// Tiled GEMM: out = (A[M,K] @ W[K,N] + bias) * oscale, bf16 WMMA, f32 acc.
// Block = 128(M)x256(N) C tile, 8 wave32 waves in 2(M)x4(N) grid, wave tile
// 64x64 = 4x4 WMMA tiles. Double-buffered 32-wide K chunks in LDS.
// AT: float (convert on stage) or __bf16 (async raw b128 copy).
// MODE: 0 -> f32 row-major [M,N]; 1 -> bf16 scatter [B,G,HG,S,D];
//       2 -> bf16 scatter [B,G,S,D]
// ---------------------------------------------------------------------------
template <int MODE, typename AT>
__global__ __launch_bounds__(256) void gemm_wmma(
    const AT* __restrict__ A, const float* __restrict__ W,
    const float* __restrict__ bias, void* __restrict__ outv,
    int M, int N, int K, float oscale) {
  __shared__ __attribute__((aligned(16))) __bf16 At[2][128][40];  // [m][k]
  __shared__ __attribute__((aligned(16))) __bf16 Wt[2][256][40];  // [n][k]

  const int tid   = threadIdx.x;
  const int wave  = tid >> 5;
  const int lane  = tid & 31;
  const int lhalf = lane >> 4;
  const int lcol  = lane & 15;
  const int wm    = wave >> 2;  // 0..1  (64 rows each)
  const int wn    = wave & 3;   // 0..3  (64 cols each)
  const int bm    = blockIdx.y * 128;
  const int bn    = blockIdx.x * 256;

  const int ar   = tid >> 1;         // 0..127 : A-tile row
  const int aseg = (tid & 1) * 16;   // half-row (16 of 32 K values)
  const int wr   = tid & 31;         // 0..31  : W-tile k row (lane-fast!)
  const int wseg = (tid >> 5) * 32;  // 32 N columns per thread

  auto stage = [&](int buf, int k0) {
    // ---- A tile (128x32 -> bf16 LDS [m][k]) ----
    if constexpr (sizeof(AT) == 4) {
      const float* p = (const float*)A + (size_t)(bm + ar) * K + k0 + aseg;
      if (k0 + 32 < K) __builtin_prefetch(p + 32, 0, 3);
#pragma unroll
      for (int half = 0; half < 2; ++half) {
        float f[8];
        *(float4*)&f[0] = *(const float4*)(p + half * 8);
        *(float4*)&f[4] = *(const float4*)(p + half * 8 + 4);
        v8bf bv8;
#pragma unroll
        for (int i = 0; i < 8; ++i) bv8[i] = (__bf16)f[i];
        *(v8bf*)&At[buf][ar][aseg + half * 8] = bv8;
      }
    } else {
      const __bf16* p = (const __bf16*)A + (size_t)(bm + ar) * K + k0 + aseg;
      const unsigned lds0 = (unsigned)(uintptr_t)&At[buf][ar][aseg];
      async_copy_b128(lds0,      (const void*)p);
      async_copy_b128(lds0 + 16, (const void*)(p + 8));
    }
    // ---- W tile (32x256 f32 -> transposed bf16 LDS [n][k]) ----
    {
      const float* p = W + (size_t)(k0 + wr) * N + bn + wseg;
      if (k0 + 32 < K) __builtin_prefetch(p + (size_t)32 * N, 0, 3);
#pragma unroll
      for (int i = 0; i < 32; i += 4) {
        float4 f = *(const float4*)(p + i);
        Wt[buf][wseg + i + 0][wr] = (__bf16)f.x;
        Wt[buf][wseg + i + 1][wr] = (__bf16)f.y;
        Wt[buf][wseg + i + 2][wr] = (__bf16)f.z;
        Wt[buf][wseg + i + 3][wr] = (__bf16)f.w;
      }
    }
  };

  v8f acc[4][4];
#pragma unroll
  for (int mt = 0; mt < 4; ++mt)
#pragma unroll
    for (int nt = 0; nt < 4; ++nt) acc[mt][nt] = vzero8();

  stage(0, 0);
  int cur = 0;
  for (int k0 = 0; k0 < K; k0 += 32) {
    if constexpr (sizeof(AT) == 2) wait_async0();  // previous async stage done
    __syncthreads();
    if (k0 + 32 < K) stage(cur ^ 1, k0 + 32);

    v16bf af[4], bf[4];
#pragma unroll
    for (int mt = 0; mt < 4; ++mt)
      af[mt] = frag_ld(&At[cur][wm * 64 + mt * 16 + lcol][0], lhalf);
#pragma unroll
    for (int nt = 0; nt < 4; ++nt)
      bf[nt] = frag_ld(&Wt[cur][wn * 64 + nt * 16 + lcol][0], lhalf);
#pragma unroll
    for (int mt = 0; mt < 4; ++mt)
#pragma unroll
      for (int nt = 0; nt < 4; ++nt)
        acc[mt][nt] = wmma_bf16(af[mt], bf[nt], acc[mt][nt]);
    cur ^= 1;
  }

  // ---- epilogue: bias, scale, scatter ----
#pragma unroll
  for (int mt = 0; mt < 4; ++mt) {
#pragma unroll
    for (int nt = 0; nt < 4; ++nt) {
      const int col = bn + wn * 64 + nt * 16 + lcol;
      const float bv = bias[col];
#pragma unroll
      for (int jj = 0; jj < 8; ++jj) {
        const int row = bm + wm * 64 + mt * 16 + lhalf * 8 + jj;
        const float val = (acc[mt][nt][jj] + bv) * oscale;
        if (MODE == 0) {
          ((float*)outv)[(size_t)row * N + col] = val;
        } else {
          const int bi = row >> 11;   // row / S
          const int si = row & 2047;  // row % S
          size_t idx;
          if (MODE == 1) {            // col -> (g,h,d), out [B,G,HG,S,D]
            const int g = col >> 9;
            const int h = (col >> 7) & 3;
            const int d = col & 127;
            idx = ((((size_t)(bi * kG + g) * kHG + h) * kS) + si) * kD + d;
          } else {                    // col -> (g,d), out [B,G,S,D]
            const int g = col >> 7;
            const int d = col & 127;
            idx = (((size_t)(bi * kG + g) * kS) + si) * kD + d;
          }
          ((__bf16*)outv)[idx] = (__bf16)val;
        }
      }
    }
  }
}

// ---------------------------------------------------------------------------
// Flash-style grouped attention (all-bf16 operands, f32 accumulation).
// Grid: (S/128, B*G*HG). Block: 256 threads (8 waves); wave owns 16 Q rows.
// K tiles staged via global_load_async_to_lds_b128 (ASYNCcnt); V transposed
// through LDS; P re-laid-out C->A through per-wave LDS; WMMA throughout.
// ---------------------------------------------------------------------------
__global__ __launch_bounds__(256) void gqa_attn(
    const __bf16* __restrict__ Qp, const __bf16* __restrict__ Kp,
    const __bf16* __restrict__ Vp, __bf16* __restrict__ ctx) {
  constexpr int BT = 64;  // K/V rows per iteration
  __shared__ __attribute__((aligned(16))) __bf16 Kt[BT][136];    // [t][d]
  __shared__ __attribute__((aligned(16))) __bf16 Vt[kD][80];     // [d][t]
  __shared__ __attribute__((aligned(16))) __bf16 Pt[8][16][80];  // per-wave P

  const int tid   = threadIdx.x;
  const int wave  = tid >> 5;
  const int lane  = tid & 31;
  const int lhalf = lane >> 4;
  const int lcol  = lane & 15;

  const int bgh = blockIdx.y;  // ((b*G+g)*HG+h)
  const int b   = bgh >> 4;
  const int g   = (bgh >> 2) & 3;
  const int h   = bgh & 3;
  const __bf16* Qbase = Qp + (size_t)bgh * kS * kD;
  const __bf16* Kbase = Kp + (size_t)(b * kG + g) * kS * kD;
  const __bf16* Vbase = Vp + (size_t)(b * kG + g) * kS * kD;
  const int q0 = blockIdx.x * 128 + wave * 16;

  // ---- Q fragments straight from global (scale pre-folded at projection) ----
  v16bf qf[4];
#pragma unroll
  for (int kc = 0; kc < 4; ++kc)
    qf[kc] = frag_ld(Qbase + (size_t)(q0 + lcol) * kD + kc * 32, lhalf);

  v8f o[8];
#pragma unroll
  for (int dt = 0; dt < 8; ++dt) o[dt] = vzero8();
  float mrun[8], lrun[8];
#pragma unroll
  for (int jj = 0; jj < 8; ++jj) { mrun[jj] = -1e30f; lrun[jj] = 0.f; }

  const int r  = tid >> 2;        // 0..63 t-row for staging
  const int c0 = (tid & 3) * 32;  // 32 d columns

  for (int t0 = 0; t0 < kS; t0 += BT) {
    // ---- K tile: async global->LDS byte copy (4 x b128 per thread) ----
    {
      const __bf16* gk = Kbase + (size_t)(t0 + r) * kD + c0;
      const unsigned lds0 = (unsigned)(uintptr_t)&Kt[r][c0];
#pragma unroll
      for (int i = 0; i < 4; ++i)
        async_copy_b128(lds0 + i * 16, (const void*)(gk + i * 8));
    }
    // ---- V tile: transpose [t][d] -> [d][t] via b16 scatter ----
    {
      const __bf16* pv = Vbase + (size_t)(t0 + r) * kD + c0;
#pragma unroll
      for (int i = 0; i < 32; i += 8) {
        __bf16 tmp[8];
        *(uint4*)tmp = *(const uint4*)(pv + i);
#pragma unroll
        for (int j = 0; j < 8; ++j) Vt[c0 + i + j][r] = tmp[j];
      }
    }
    wait_async0();     // our own async copies landed in LDS
    __syncthreads();   // publish to all waves

    // ---- scores: S = Q @ K^T, 4 tiles of 16x16 ----
    v8f sc[4];
#pragma unroll
    for (int j = 0; j < 4; ++j) {
      v8f a = vzero8();
#pragma unroll
      for (int kc = 0; kc < 4; ++kc) {
        v16bf bfr = frag_ld(&Kt[j * 16 + lcol][kc * 32], lhalf);
        a = wmma_bf16(qf[kc], bfr, a);
      }
      sc[j] = a;
    }

    // ---- online softmax (C-layout rows: row = lhalf*8 + jj) ----
#pragma unroll
    for (int jj = 0; jj < 8; ++jj) {
      float mx = fmaxf(fmaxf(sc[0][jj], sc[1][jj]), fmaxf(sc[2][jj], sc[3][jj]));
#pragma unroll
      for (int msk = 8; msk >= 1; msk >>= 1) mx = fmaxf(mx, __shfl_xor(mx, msk, 32));
      const float mnew  = fmaxf(mrun[jj], mx);
      const float alpha = __expf(mrun[jj] - mnew);
      float ps = 0.f;
#pragma unroll
      for (int j = 0; j < 4; ++j) {
        const float e = __expf(sc[j][jj] - mnew);
        sc[j][jj] = e;
        ps += e;
      }
#pragma unroll
      for (int msk = 8; msk >= 1; msk >>= 1) ps += __shfl_xor(ps, msk, 32);
      lrun[jj] = lrun[jj] * alpha + ps;
      mrun[jj] = mnew;
#pragma unroll
      for (int dt = 0; dt < 8; ++dt) o[dt][jj] *= alpha;
    }

    // ---- re-layout P (C layout -> A fragments) via per-wave LDS ----
#pragma unroll
    for (int j = 0; j < 4; ++j)
#pragma unroll
      for (int jj = 0; jj < 8; ++jj)
        Pt[wave][lhalf * 8 + jj][j * 16 + lcol] = (__bf16)sc[j][jj];
    __syncthreads();

    // ---- O += P @ V ----
#pragma unroll
    for (int dt = 0; dt < 8; ++dt) {
      v8f a = o[dt];
#pragma unroll
      for (int tc = 0; tc < 2; ++tc) {
        v16bf pa = frag_ld(&Pt[wave][lcol][tc * 32], lhalf);
        v16bf vb = frag_ld(&Vt[dt * 16 + lcol][tc * 32], lhalf);
        a = wmma_bf16(pa, vb, a);
      }
      o[dt] = a;
    }
    __syncthreads();  // protect Kt/Vt/Pt before next tile
  }

  // ---- normalize, write ctx bf16 in [B,S,(g,h,d)] layout ----
#pragma unroll
  for (int jj = 0; jj < 8; ++jj) {
    const int srow = q0 + lhalf * 8 + jj;
    const float inv = 1.f / lrun[jj];
    const size_t base = ((size_t)(b * kS + srow) * kE) + (size_t)(g * kHG + h) * kD;
#pragma unroll
    for (int dt = 0; dt < 8; ++dt)
      ctx[base + dt * 16 + lcol] = (__bf16)(o[dt][jj] * inv);
  }
}

// ---------------------------------------------------------------------------
extern "C" void kernel_launch(void* const* d_in, const int* in_sizes, int n_in,
                              void* d_out, int out_size, void* d_ws, size_t ws_size,
                              hipStream_t stream) {
  (void)in_sizes; (void)n_in; (void)out_size; (void)ws_size;
  const float* q  = (const float*)d_in[0];
  const float* k  = (const float*)d_in[1];
  const float* v  = (const float*)d_in[2];
  const float* Wq = (const float*)d_in[3];
  const float* bq = (const float*)d_in[4];
  const float* Wk = (const float*)d_in[5];
  const float* bk = (const float*)d_in[6];
  const float* Wv = (const float*)d_in[7];
  const float* bv = (const float*)d_in[8];
  const float* Wo = (const float*)d_in[9];
  const float* bo = (const float*)d_in[10];
  float* out = (float*)d_out;

  // workspace (bf16): Qp [B,G,HG,S,D] | Kp [B,G,S,D] | Vp [B,G,S,D] | ctx [B,S,E]
  __bf16* Qp   = (__bf16*)d_ws;
  __bf16* Kp   = Qp + (size_t)16777216;  // 4*4*4*2048*128
  __bf16* Vp   = Kp + (size_t)4194304;   // 4*4*2048*128
  __bf16* ctxp = Vp + (size_t)4194304;

  const int M = kB * kS;  // 8192
  const float qscale = 0.08838834764831845f;  // 1/sqrt(128)
  dim3 blk(256);

  gemm_wmma<1, float><<<dim3(kE / 256, M / 128), blk, 0, stream>>>(
      q, Wq, bq, Qp, M, kE, kE, qscale);
  gemm_wmma<2, float><<<dim3((kG * kD) / 256, M / 128), blk, 0, stream>>>(
      k, Wk, bk, Kp, M, kG * kD, kE, 1.0f);
  gemm_wmma<2, float><<<dim3((kG * kD) / 256, M / 128), blk, 0, stream>>>(
      v, Wv, bv, Vp, M, kG * kD, kE, 1.0f);
  gqa_attn<<<dim3(kS / 128, kB * kG * kHG), blk, 0, stream>>>(Qp, Kp, Vp, ctxp);
  gemm_wmma<0, __bf16><<<dim3(kE / 256, M / 128), blk, 0, stream>>>(
      ctxp, Wo, bo, out, M, kE, kE, 1.0f);
}
